// CausalAttention_79431125172811
// MI455X (gfx1250) — matrix operand order repaired and checked by reading, not compiled, not run on hardware
//
#include <hip/hip_runtime.h>
#include <hip/hip_bf16.h>
#include <stdint.h>

// ---------------- CDNA5 (gfx1250) causal attention ----------------
// Pipeline: f32->bf16 convert; QKV bf16 WMMA GEMMs (V written transposed);
// masked/scaled scores GEMM (causal tile-skip); causal softmax; P@V GEMM.
// GEMM staging: global_load_async_to_lds_b128, double-buffered with a
// statically unrolled ping-pong (KT is always even), ASYNCcnt-synchronized.
// Matmuls: v_wmma_f32_16x16x32_bf16 with fp32 accumulation.

typedef __attribute__((ext_vector_type(16))) __bf16        v16bf;
typedef __attribute__((ext_vector_type(8)))  float         v8f;
typedef __attribute__((ext_vector_type(4)))  unsigned int  v4u;
typedef __attribute__((ext_vector_type(4)))  float         v4f;
typedef __attribute__((ext_vector_type(4)))  unsigned short v4us;

#define BM 128
#define BN 128
#define BK 32
#define LDSK 40   // padded LDS row stride in bf16 elements (80B: conflict-free, 16B aligned)

__device__ __forceinline__ float bf2f(unsigned short h) {
  unsigned int u = ((unsigned int)h) << 16;
  float f; __builtin_memcpy(&f, &u, 4); return f;
}
__device__ __forceinline__ unsigned short f2bf(float f) {
  unsigned int u; __builtin_memcpy(&u, &f, 4);
  u += 0x7FFFu + ((u >> 16) & 1u);          // round-to-nearest-even
  return (unsigned short)(u >> 16);
}

// Low 32 bits of a flat shared-memory address == LDS byte offset.
__device__ __forceinline__ unsigned int lds_addr(const void* p) {
  return (unsigned int)(uintptr_t)p;
}

// Async-copy 32B (two b128) global -> LDS. INST_OFFSET applies to both the
// LDS and global address per ISA pseudocode, so one dst/src pair serves both.
__device__ __forceinline__ void async_copy_seg(const unsigned short* src, unsigned int dst) {
  unsigned long long ga = (unsigned long long)(uintptr_t)src;
  asm volatile("global_load_async_to_lds_b128 %0, %1, off"
               :: "v"(dst), "v"(ga) : "memory");
  asm volatile("global_load_async_to_lds_b128 %0, %1, off offset:16"
               :: "v"(dst), "v"(ga) : "memory");
}

union FragU { v16bf v; v4u q[2]; };

// A-matrix 16x32 bf16 fragment (ISA 7.12.2): lane<16 -> M=lane, K{0..7,16..23};
// lane>=16 -> M=lane-16, K{8..15,24..31}. LDS tile is [row][k], row stride LDSK.
__device__ __forceinline__ v16bf load_afrag(const unsigned short* As, int row0, int lane) {
  int r  = row0 + (lane & 15);
  int kb = (lane >> 4) << 3;                  // 0 or 8
  const unsigned short* p = As + r * LDSK + kb;
  FragU f;
  f.q[0] = *(const v4u*)(p);
  f.q[1] = *(const v4u*)(p + 16);
  return f.v;
}
// B-matrix 32x16 bf16 fragment: lane<16 -> N=lane, K{0..7,16..23}; lane>=16 -> N=lane-16,
// K{8..15,24..31}. LDS tile stored N-major [n][k], row stride LDSK.
__device__ __forceinline__ v16bf load_bfrag(const unsigned short* Bs, int col0, int lane) {
  int n  = col0 + (lane & 15);
  int kb = (lane >> 4) << 3;                  // 0 or 8
  const unsigned short* p = Bs + n * LDSK + kb;
  FragU f;
  f.q[0] = *(const v4u*)(p);
  f.q[1] = *(const v4u*)(p + 16);
  return f.v;
}

// One BK=32 step: 4 A-frags x 2 B-frags -> 8 WMMAs on a fixed LDS buffer
// (static buffer => ds_load offsets fold into immediates).
__device__ __forceinline__ void compute_step(const unsigned short* As,
                                             const unsigned short* Bs,
                                             v8f (&acc)[4][2],
                                             int wm, int wn, int lane) {
  v16bf af[4], bfr[2];
#pragma unroll
  for (int ms = 0; ms < 4; ++ms) af[ms] = load_afrag(As, wm + ms * 16, lane);
#pragma unroll
  for (int ns = 0; ns < 2; ++ns) bfr[ns] = load_bfrag(Bs, wn + ns * 16, lane);
#pragma unroll
  for (int ms = 0; ms < 4; ++ms)
#pragma unroll
    for (int ns = 0; ns < 2; ++ns)
      acc[ms][ns] = __builtin_amdgcn_wmma_f32_16x16x32_bf16(
          false, af[ms], false, bfr[ns], (short)0, acc[ms][ns], false, false);
}

// MODE 0: C = A*B^T (bf16 out, [m][n])         (QKV: Q, K)
// MODE 1: C = A*B^T written transposed as Vt[b*1024+n][2048]+s   (QKV: V)
// MODE 2: scores: scale + causal mask, bf16 out, skip tiles above diagonal
// MODE 3: P@V: f32 out, K-loop truncated at causal limit
template<int MODE>
__global__ __launch_bounds__(256)
void gemm_bf16_wmma(const unsigned short* __restrict__ A,
                    const unsigned short* __restrict__ B,
                    void* __restrict__ Cv,
                    int M, int N, int K,
                    int lda, int ldb, int ldc,
                    float scale,
                    size_t strideAz, size_t strideBz, size_t strideCz)
{
  __shared__ unsigned short lds[2][2][BM * LDSK];   // [buf][A/B][tile] = 40KB

  const int bM = blockIdx.y * BM;
  const int bN = blockIdx.x * BN;

  if (MODE == 2) {
    if (bN > bM + (BM - 1)) return;   // tile fully above causal diagonal: skip
  }

  A += (size_t)blockIdx.z * strideAz;
  B += (size_t)blockIdx.z * strideBz;

  int Keff = K;
  if (MODE == 3) { int lim = bM + BM; Keff = (lim < K) ? lim : K; }  // causal K-limit
  const int KT = Keff / BK;   // always even: K=1024 -> 32; causal limit mult. of 128 -> mult. of 4

  const int t    = threadIdx.x;
  const int lane = t & 31;
  const int wid  = t >> 5;
  const int wm   = (wid >> 2) * 64;   // wave M offset (2 waves in M)
  const int wn   = (wid & 3) * 32;    // wave N offset (4 waves in N)

  // per-thread staging segment: 32B of one tile row
  const int srow = t >> 1;
  const int scol = (t & 1) << 4;
  const unsigned short* gA = A + (size_t)(bM + srow) * lda + scol;
  const unsigned short* gB = B + (size_t)(bN + srow) * ldb + scol;
  const unsigned int dA0 = lds_addr(&lds[0][0][srow * LDSK + scol]);
  const unsigned int dB0 = lds_addr(&lds[0][1][srow * LDSK + scol]);
  const unsigned int dA1 = lds_addr(&lds[1][0][srow * LDSK + scol]);
  const unsigned int dB1 = lds_addr(&lds[1][1][srow * LDSK + scol]);

  // prefetch tile 0 -> buf0
  async_copy_seg(gA, dA0);
  async_copy_seg(gB, dB0);

  v8f acc[4][2] = {};

  for (int kt = 0; kt < KT; kt += 2) {
    // ---- even step: consume buf0, prefetch tile kt+1 -> buf1 (kt+1 < KT always)
    asm volatile("s_wait_asynccnt 0x0" ::: "memory");
    asm volatile("s_wait_dscnt 0x0" ::: "memory");
    __syncthreads();
    async_copy_seg(gA + (kt + 1) * BK, dA1);
    async_copy_seg(gB + (kt + 1) * BK, dB1);
    compute_step(&lds[0][0][0], &lds[0][1][0], acc, wm, wn, lane);

    // ---- odd step: consume buf1, prefetch tile kt+2 -> buf0
    asm volatile("s_wait_asynccnt 0x0" ::: "memory");
    asm volatile("s_wait_dscnt 0x0" ::: "memory");
    __syncthreads();
    if (kt + 2 < KT) {
      async_copy_seg(gA + (kt + 2) * BK, dA0);
      async_copy_seg(gB + (kt + 2) * BK, dB0);
    }
    compute_step(&lds[1][0][0], &lds[1][1][0], acc, wm, wn, lane);
  }

  // Epilogue. C/D layout: VGPR r, lane<16 -> M=r, N=lane; lane>=16 -> M=r+8.
  const int mofs = wm + ((lane >> 4) << 3);
  const int nofs = wn + (lane & 15);
  (void)scale; (void)M; (void)N;
#pragma unroll
  for (int ms = 0; ms < 4; ++ms) {
#pragma unroll
    for (int ns = 0; ns < 2; ++ns) {
#pragma unroll
      for (int r = 0; r < 8; ++r) {
        int row = bM + mofs + ms * 16 + r;
        int col = bN + nofs + ns * 16;
        float v = acc[ms][ns][r];
        if (MODE == 0) {
          ((unsigned short*)Cv)[(size_t)blockIdx.z * strideCz + (size_t)row * ldc + col] = f2bf(v);
        } else if (MODE == 1) {
          int b = row >> 11, s = row & 2047;       // S=2048
          ((unsigned short*)Cv)[((size_t)(b * 1024 + col)) * 2048 + s] = f2bf(v);  // V^T
        } else if (MODE == 2) {
          unsigned short o = (col <= row) ? f2bf(v * scale) : (unsigned short)0xFF80; // -inf
          ((unsigned short*)Cv)[(size_t)blockIdx.z * strideCz + (size_t)row * ldc + col] = o;
        } else {
          ((float*)Cv)[(size_t)blockIdx.z * strideCz + (size_t)row * ldc + col] = v;
        }
      }
    }
  }
}

__global__ __launch_bounds__(256)
void softmax_causal(unsigned short* __restrict__ P, int S)
{
  const int rowg = blockIdx.x;              // 0 .. B*S-1
  const int i = rowg & (S - 1);
  unsigned short* row = P + (size_t)rowg * S;
  const int n = i + 1;
  const int t = threadIdx.x;

  __shared__ float red[8];

  float m = -3.4e38f;
  for (int j = t; j < n; j += 256) m = fmaxf(m, bf2f(row[j]));
#pragma unroll
  for (int off = 16; off > 0; off >>= 1) m = fmaxf(m, __shfl_xor(m, off, 32));
  if ((t & 31) == 0) red[t >> 5] = m;
  __syncthreads();
  if (t < 32) {
    float v = (t < 8) ? red[t] : -3.4e38f;
#pragma unroll
    for (int off = 4; off > 0; off >>= 1) v = fmaxf(v, __shfl_xor(v, off, 32));
    if (t == 0) red[0] = v;
  }
  __syncthreads();
  m = red[0];
  __syncthreads();

  float s = 0.f;
  for (int j = t; j < n; j += 256) s += __expf(bf2f(row[j]) - m);
#pragma unroll
  for (int off = 16; off > 0; off >>= 1) s += __shfl_xor(s, off, 32);
  if ((t & 31) == 0) red[t >> 5] = s;
  __syncthreads();
  if (t < 32) {
    float v = (t < 8) ? red[t] : 0.f;
#pragma unroll
    for (int off = 4; off > 0; off >>= 1) v += __shfl_xor(v, off, 32);
    if (t == 0) red[0] = v;
  }
  __syncthreads();
  const float inv = 1.f / red[0];

  for (int j = t; j < n; j += 256) row[j] = f2bf(__expf(bf2f(row[j]) - m) * inv);
  for (int j = n + t; j < S; j += 256) row[j] = 0;   // zero masked tail for P@V
}

__global__ __launch_bounds__(256)
void cvt_f32_bf16(const float* __restrict__ src, unsigned short* __restrict__ dst, long long n4)
{
  long long i = (long long)blockIdx.x * blockDim.x + threadIdx.x;
  long long stride = (long long)gridDim.x * blockDim.x;
  for (; i < n4; i += stride) {
    v4f f = ((const v4f*)src)[i];
    v4us o;
    o.x = f2bf(f.x); o.y = f2bf(f.y); o.z = f2bf(f.z); o.w = f2bf(f.w);
    ((v4us*)dst)[i] = o;
  }
}

extern "C" void kernel_launch(void* const* d_in, const int* in_sizes, int n_in,
                              void* d_out, int out_size, void* d_ws, size_t ws_size,
                              hipStream_t stream)
{
  const float* x  = (const float*)d_in[0];
  const float* Wq = (const float*)d_in[1];
  const float* Wk = (const float*)d_in[2];
  const float* Wv = (const float*)d_in[3];
  float* out = (float*)d_out;

  const int Bn = 4, S = 2048, D = 1024;
  const size_t MSZ = (size_t)Bn * S * D;     // 8,388,608 elems (x / Q / K / Vt)
  const size_t WSZ = (size_t)D * D;          // 1,048,576 elems per weight
  const size_t PSZ = (size_t)Bn * S * S;     // 16,777,216 elems (scores/P)

  unsigned short* XB  = (unsigned short*)d_ws;
  unsigned short* WBq = XB + MSZ;
  unsigned short* WBk = WBq + WSZ;
  unsigned short* WBv = WBk + WSZ;
  unsigned short* Qb  = WBv + WSZ;
  unsigned short* Kb  = Qb + MSZ;
  unsigned short* Vt  = Kb + MSZ;            // stored [b][n=1024][s=2048]
  unsigned short* Pb  = Vt + MSZ;            // scores -> probs in place, bf16

  const size_t need = 2u * (MSZ * 4 + WSZ * 3 + PSZ);   // ~102 MB
  if (ws_size < need) return;

  // f32 -> bf16 converts
  cvt_f32_bf16<<<2048, 256, 0, stream>>>(x,  XB,  (long long)(MSZ / 4));
  cvt_f32_bf16<<<512,  256, 0, stream>>>(Wq, WBq, (long long)(WSZ / 4));
  cvt_f32_bf16<<<512,  256, 0, stream>>>(Wk, WBk, (long long)(WSZ / 4));
  cvt_f32_bf16<<<512,  256, 0, stream>>>(Wv, WBv, (long long)(WSZ / 4));

  // QKV projections: y = x @ W^T ; W is [n][k] so B staging is contiguous rows
  dim3 gQKV(D / BN, (Bn * S) / BM, 1);       // (8, 64)
  gemm_bf16_wmma<0><<<gQKV, 256, 0, stream>>>(XB, WBq, Qb, Bn*S, D, D, D, D, D, 1.f, 0, 0, 0);
  gemm_bf16_wmma<0><<<gQKV, 256, 0, stream>>>(XB, WBk, Kb, Bn*S, D, D, D, D, D, 1.f, 0, 0, 0);
  gemm_bf16_wmma<1><<<gQKV, 256, 0, stream>>>(XB, WBv, Vt, Bn*S, D, D, D, D, D, 1.f, 0, 0, 0);

  // scores = (Q @ K^T) * 1/sqrt(D), causal-masked; K rows are [j][d] -> contiguous
  dim3 gS(S / BN, S / BM, Bn);               // (16, 16, 4); upper tiles skipped
  gemm_bf16_wmma<2><<<gS, 256, 0, stream>>>(Qb, Kb, Pb, S, S, D, D, D, S, 0.03125f,
                                            (size_t)S * D, (size_t)S * D, (size_t)S * S);

  softmax_causal<<<Bn * S, 256, 0, stream>>>(Pb, S);

  // out = P @ V ; B operand is Vt [n][s] -> contiguous, causal K-limit
  dim3 gO(D / BN, S / BM, Bn);               // (8, 16, 4)
  gemm_bf16_wmma<3><<<gO, 256, 0, stream>>>(Pb, Vt, out, S, D, S, S, S, D, 1.f,
                                            (size_t)S * S, (size_t)D * S, (size_t)S * D);
}
